// ODELSTMCell_3805341024598
// MI455X (gfx1250) — matrix operand order repaired and checked by reading, not compiled, not run on hardware
//
#include <hip/hip_runtime.h>
#include <hip/hip_bf16.h>
#include <stdint.h>

// ---------------------------------------------------------------------------
// ODE-LSTM on MI455X (gfx1250): bf16 WMMA GEMM pipeline with fused epilogues.
// ---------------------------------------------------------------------------

typedef __attribute__((ext_vector_type(16))) __bf16 v16bf;
typedef __attribute__((ext_vector_type(8)))  float  v8f;

__device__ __forceinline__ unsigned short f2bf(float f) {
  unsigned int u = __float_as_uint(f);
  unsigned int r = (u + 0x7FFFu + ((u >> 16) & 1u)) >> 16;  // round-nearest-even
  return (unsigned short)r;
}
__device__ __forceinline__ float sigmoid_f(float x) { return 1.0f / (1.0f + __expf(-x)); }
__device__ __forceinline__ float tanh_f(float x) {
  float e = __expf(-2.0f * x);
  return (1.0f - e) / (1.0f + e);
}

#define BM 128
#define BN 64
#define BK 64
#define LDSA 72   // bf16 elems per A row in LDS (pad 8 -> 144B rows, 16B aligned)
#define LDSB 72
#define LDSC 68   // f32 elems per C row in LDS epilogue
#define SMEM_BYTES 34816  // max(128*72*2 + 64*72*2 = 27648, 128*68*4 = 34816)

// MODE 0: out f32 = acc + bias                (MLP second layer -> k)
// MODE 1: out bf16 = tanh(acc + bias)        (MLP first layer -> u)
// MODE 2: fused LSTM epilogue (gate-interleaved N), writes h_bf16, h_f32, c
template <int MODE>
__global__ __launch_bounds__(256) void gemm_bf16_wmma(
    const unsigned short* __restrict__ A, int lda,
    const unsigned short* __restrict__ W, int ldw,
    const float* __restrict__ bias,
    int K, int N,
    float* __restrict__ outF,
    unsigned short* __restrict__ outBF, int ldob,
    const float* __restrict__ cin,
    float* __restrict__ cout,
    float* __restrict__ hf32)
{
  __shared__ __align__(16) char smem[SMEM_BYTES];
  unsigned short* ldsA = (unsigned short*)smem;      // [128][72]
  unsigned short* ldsB = ldsA + BM * LDSA;           // [64][72]

  const int tid  = threadIdx.x;
  const int lane = tid & 31;
  const int wid  = tid >> 5;
  const int waveM = wid >> 1;       // 0..3 -> 32-row slab
  const int waveN = wid & 1;        // 0..1 -> 32-col slab
  const int laneHalf = lane >> 4;   // 0/1
  const int lane15   = lane & 15;
  const int bx = blockIdx.x;        // N tile
  const int by = blockIdx.y;        // M tile

  v8f acc[2][2];
#pragma unroll
  for (int i = 0; i < 2; ++i)
#pragma unroll
    for (int j = 0; j < 2; ++j)
      acc[i][j] = (v8f){0.f, 0.f, 0.f, 0.f, 0.f, 0.f, 0.f, 0.f};

  const unsigned short* Ag = A + (size_t)(by * BM) * lda;
  const unsigned short* Wg = W + (size_t)(bx * BN) * ldw;

  for (int kt = 0; kt < K; kt += BK) {
    // ---- stage global -> regs (b128) ----
    uint4 ar[4], br[2];
#pragma unroll
    for (int i = 0; i < 4; ++i) {
      int e = (tid + i * 256) * 8;          // elem index within 128x64 tile
      int r = e >> 6, col = e & 63;
      ar[i] = *(const uint4*)(Ag + (size_t)r * lda + kt + col);
    }
#pragma unroll
    for (int i = 0; i < 2; ++i) {
      int e = (tid + i * 256) * 8;          // within 64x64 tile
      int r = e >> 6, col = e & 63;
      br[i] = *(const uint4*)(Wg + (size_t)r * ldw + kt + col);
    }
    __syncthreads();   // prior iteration's fragment reads complete
#pragma unroll
    for (int i = 0; i < 4; ++i) {
      int e = (tid + i * 256) * 8;
      int r = e >> 6, col = e & 63;
      *(uint4*)&ldsA[r * LDSA + col] = ar[i];
    }
#pragma unroll
    for (int i = 0; i < 2; ++i) {
      int e = (tid + i * 256) * 8;
      int r = e >> 6, col = e & 63;
      *(uint4*)&ldsB[r * LDSB + col] = br[i];
    }
    __syncthreads();

    // ---- two K=32 WMMA steps per LDS tile ----
#pragma unroll
    for (int ks = 0; ks < 2; ++ks) {
      v16bf aF[2], bF[2];
#pragma unroll
      for (int tm = 0; tm < 2; ++tm) {
        // A 16x32 bf16 layout: lanes 0-15 rows M, K 0..7 + 16..23; lanes 16-31: K 8..15 + 24..31
        int r  = waveM * 32 + tm * 16 + lane15;
        int kb = ks * 32 + laneHalf * 8;
        union { uint4 u[2]; v16bf v; } f;
        f.u[0] = *(const uint4*)&ldsA[r * LDSA + kb];
        f.u[1] = *(const uint4*)&ldsA[r * LDSA + kb + 16];
        aF[tm] = f.v;
      }
#pragma unroll
      for (int tn = 0; tn < 2; ++tn) {
        // B 32x16 bf16 layout: lane n holds column n (= W row n); lanes 0-15: K 0..7 + 16..23, lanes 16-31: K 8..15 + 24..31
        int r  = waveN * 32 + tn * 16 + lane15;
        int kb = ks * 32 + laneHalf * 8;
        union { uint4 u[2]; v16bf v; } f;
        f.u[0] = *(const uint4*)&ldsB[r * LDSB + kb];
        f.u[1] = *(const uint4*)&ldsB[r * LDSB + kb + 16];
        bF[tn] = f.v;
      }
#pragma unroll
      for (int tm = 0; tm < 2; ++tm)
#pragma unroll
        for (int tn = 0; tn < 2; ++tn)
          acc[tm][tn] = __builtin_amdgcn_wmma_f32_16x16x32_bf16(
              false, aF[tm], false, bF[tn], (short)0, acc[tm][tn], false, false);
    }
  }

  if (MODE == 2) {
    // ---- fused LSTM epilogue: N is gate-interleaved (n = h*4 + gate) ----
    __syncthreads();                     // done reading A/B LDS tiles
    float* Cb = (float*)smem;            // [128][68]
#pragma unroll
    for (int tm = 0; tm < 2; ++tm)
#pragma unroll
      for (int tn = 0; tn < 2; ++tn)
#pragma unroll
        for (int v = 0; v < 8; ++v) {
          int r  = waveM * 32 + tm * 16 + (laneHalf << 3) + v;  // C layout: lanes16-31 -> M+8
          int cc = waveN * 32 + tn * 16 + lane15;
          Cb[r * LDSC + cc] = acc[tm][tn][v];
        }
    __syncthreads();
    const int H = N >> 2;                // 1024
    for (int idx = tid; idx < BM * 16; idx += 256) {
      int m = idx >> 4, hl = idx & 15;
      size_t gRow = (size_t)(by * BM + m);
      int hG = bx * 16 + hl;
      int nB = bx * BN + hl * 4;
      float gi = Cb[m * LDSC + hl * 4 + 0] + bias[nB + 0];
      float gf = Cb[m * LDSC + hl * 4 + 1] + bias[nB + 1];
      float gg = Cb[m * LDSC + hl * 4 + 2] + bias[nB + 2];
      float go = Cb[m * LDSC + hl * 4 + 3] + bias[nB + 3];
      float iv = sigmoid_f(gi), fv = sigmoid_f(gf);
      float gv = tanh_f(gg),    ov = sigmoid_f(go);
      float cold = cin[gRow * H + hG];
      float cn = fv * cold + iv * gv;
      float hn = ov * tanh_f(cn);
      cout[gRow * H + hG] = cn;
      hf32[gRow * H + hG] = hn;
      outBF[gRow * (size_t)ldob + hG] = f2bf(hn);
    }
  } else {
    // ---- direct epilogues ----
#pragma unroll
    for (int tm = 0; tm < 2; ++tm)
#pragma unroll
      for (int tn = 0; tn < 2; ++tn)
#pragma unroll
        for (int v = 0; v < 8; ++v) {
          int rL = waveM * 32 + tm * 16 + (laneHalf << 3) + v;
          int cL = waveN * 32 + tn * 16 + lane15;
          size_t rG = (size_t)(by * BM + rL);
          int cG = bx * BN + cL;
          float val = acc[tm][tn][v] + bias[cG];
          if (MODE == 0) outF[rG * (size_t)N + cG] = val;
          else           outBF[rG * (size_t)ldob + cG] = f2bf(tanh_f(val));
        }
  }
}

// ------------------------- prep / elementwise kernels ----------------------

// Gate-interleaved concat weight: row j = h*4+g holds [w_ih[g*H+h,:512] , w_hh[g*H+h,:1024]] (bf16)
__global__ void prep_wint_k(const float* __restrict__ w_ih, const float* __restrict__ w_hh,
                            unsigned short* __restrict__ Wint) {
  int idx = blockIdx.x * 256 + threadIdx.x;
  int j = idx / 1536, k = idx - j * 1536;
  int g = j & 3, hh = j >> 2;
  float v = (k < 512) ? w_ih[(size_t)(g * 1024 + hh) * 512 + k]
                      : w_hh[(size_t)(g * 1024 + hh) * 1024 + (k - 512)];
  Wint[idx] = f2bf(v);
}

__global__ void prep_bias_k(const float* __restrict__ b_ih, const float* __restrict__ b_hh,
                            float* __restrict__ bint) {
  int j = blockIdx.x * 256 + threadIdx.x;
  int g = j & 3, hh = j >> 2;
  bint[j] = b_ih[g * 1024 + hh] + b_hh[g * 1024 + hh];
}

__global__ void f32_to_bf16_k(const float* __restrict__ s, unsigned short* __restrict__ d, int n) {
  int i = blockIdx.x * 256 + threadIdx.x;
  if (i < n) d[i] = f2bf(s[i]);
}

__global__ void pack_rows_k(const float* __restrict__ s, unsigned short* __restrict__ d,
                            int rowLen, int dstStride, int dstOff) {
  int i = blockIdx.x * 256 + threadIdx.x;
  int b = i / rowLen, k = i - b * rowLen;
  d[(size_t)b * dstStride + dstOff + k] = f2bf(s[i]);
}

__global__ void rk4_update_k(const float* __restrict__ h, const float* __restrict__ kv,
                             const float* __restrict__ ts, float* __restrict__ acc,
                             unsigned short* __restrict__ ybf,
                             float alpha, float wk, float accScale, int H) {
  int i = blockIdx.x * 256 + threadIdx.x;
  int b = i / H;
  float dt = ts[2 * b + 1] - ts[2 * b];
  float k = kv[i];
  acc[i] = accScale * acc[i] + wk * k;
  ybf[i] = f2bf(h[i] + alpha * dt * k);
}

__global__ void rk4_final_k(const float* __restrict__ h, const float* __restrict__ k4,
                            const float* __restrict__ acc, const float* __restrict__ ts,
                            float* __restrict__ ht, int H) {
  int i = blockIdx.x * 256 + threadIdx.x;
  int b = i / H;
  float dt = ts[2 * b + 1] - ts[2 * b];
  ht[i] = h[i] + (dt * (1.0f / 6.0f)) * (acc[i] + k4[i]);
}

// ------------------------------- launcher ----------------------------------

extern "C" void kernel_launch(void* const* d_in, const int* in_sizes, int n_in,
                              void* d_out, int out_size, void* d_ws, size_t ws_size,
                              hipStream_t stream) {
  (void)in_sizes; (void)n_in; (void)out_size; (void)ws_size;
  const float* x    = (const float*)d_in[0];
  const float* h0   = (const float*)d_in[1];
  const float* c0   = (const float*)d_in[2];
  const float* ts   = (const float*)d_in[3];
  const float* w_ih = (const float*)d_in[4];
  const float* w_hh = (const float*)d_in[5];
  const float* b_ih = (const float*)d_in[6];
  const float* b_hh = (const float*)d_in[7];
  const float* w1   = (const float*)d_in[8];
  const float* b1   = (const float*)d_in[9];
  const float* w2   = (const float*)d_in[10];
  const float* b2   = (const float*)d_in[11];

  const int B = 4096, IN = 512, H = 1024, G = 4 * H, KC = IN + H;  // G=4096, KC=1536

  char* ws = (char*)d_ws;
  size_t off = 0;
  auto carve = [&](size_t bytes) -> char* {
    char* p = ws + off;
    off = (off + bytes + 255) & ~(size_t)255;
    return p;
  };
  unsigned short* Wint  = (unsigned short*)carve((size_t)G * KC * 2);   // 12.6 MB
  unsigned short* w1b   = (unsigned short*)carve((size_t)H * H * 2);    // 2 MB
  unsigned short* w2b   = (unsigned short*)carve((size_t)H * H * 2);    // 2 MB
  float*          bint  = (float*)carve((size_t)G * 4);
  unsigned short* Acat1 = (unsigned short*)carve((size_t)B * KC * 2);   // 12.6 MB
  unsigned short* Acat2 = (unsigned short*)carve((size_t)B * KC * 2);   // 12.6 MB
  float*          c1    = (float*)carve((size_t)B * H * 4);             // 16.8 MB
  float*          h2    = (float*)carve((size_t)B * H * 4);             // 16.8 MB
  unsigned short* ybf   = (unsigned short*)carve((size_t)B * H * 2);    // 8.4 MB
  unsigned short* ubf   = (unsigned short*)carve((size_t)B * H * 2);    // 8.4 MB
  float*          kbuf  = (float*)carve((size_t)B * H * 4);             // 16.8 MB
  float*          accb  = (float*)carve((size_t)B * H * 4);             // 16.8 MB

  float* outHt = (float*)d_out;
  float* outC  = (float*)d_out + (size_t)B * H;

  // ---- prep: bf16 conversion + gate interleave ----
  prep_wint_k<<<(G * KC) / 256, 256, 0, stream>>>(w_ih, w_hh, Wint);
  f32_to_bf16_k<<<(H * H) / 256, 256, 0, stream>>>(w1, w1b, H * H);
  f32_to_bf16_k<<<(H * H) / 256, 256, 0, stream>>>(w2, w2b, H * H);
  prep_bias_k<<<G / 256, 256, 0, stream>>>(b_ih, b_hh, bint);
  pack_rows_k<<<(B * IN) / 256, 256, 0, stream>>>(x, Acat1, IN, KC, 0);
  pack_rows_k<<<(B * IN) / 256, 256, 0, stream>>>(x, Acat2, IN, KC, 0);
  pack_rows_k<<<(B * H) / 256, 256, 0, stream>>>(h0, Acat1, H, KC, IN);

  dim3 blk(256);
  dim3 gLSTM(G / BN, B / BM);  // 64 x 32
  dim3 gRK(H / BN, B / BM);    // 16 x 32

  // ---- LSTM step 1: gates = [x|h0] @ Wint^T ; writes h1(bf16)->Acat2 h-slot, c1 ----
  gemm_bf16_wmma<2><<<gLSTM, blk, 0, stream>>>(Acat1, KC, Wint, KC, bint, KC, G,
                                               nullptr, Acat2 + IN, KC, c0, c1, h2);
  // ---- LSTM step 2: writes h2(bf16)->ybf, h2(f32), c2 -> d_out[BH..2BH) ----
  gemm_bf16_wmma<2><<<gLSTM, blk, 0, stream>>>(Acat2, KC, Wint, KC, bint, KC, G,
                                               nullptr, ybf, H, c1, outC, h2);

  // ---- RK4: f(y) = tanh(y@w1^T+b1)@w2^T+b2, 4 stages ----
  const float alphas[3] = {0.5f, 0.5f, 1.0f};
  const float wks[3]    = {1.0f, 2.0f, 2.0f};
  for (int s = 0; s < 4; ++s) {
    gemm_bf16_wmma<1><<<gRK, blk, 0, stream>>>(ybf, H, w1b, H, b1, H, H,
                                               nullptr, ubf, H, nullptr, nullptr, nullptr);
    gemm_bf16_wmma<0><<<gRK, blk, 0, stream>>>(ubf, H, w2b, H, b2, H, H,
                                               kbuf, nullptr, 0, nullptr, nullptr, nullptr);
    if (s < 3)
      rk4_update_k<<<(B * H) / 256, 256, 0, stream>>>(h2, kbuf, ts, accb, ybf,
                                                      alphas[s], wks[s],
                                                      s == 0 ? 0.0f : 1.0f, H);
    else
      rk4_final_k<<<(B * H) / 256, 256, 0, stream>>>(h2, kbuf, accb, ts, outHt, H);
  }
}
